// SAS_7267084665097
// MI455X (gfx1250) — matrix operand order repaired and checked by reading, not compiled, not run on hardware
//
#include <hip/hip_runtime.h>
#include <math.h>

// ---------------------------------------------------------------------------
// SAS forward for gfx1250 (MI455X). Pointwise convs run as WMMA f16 GEMMs
// (f32 accumulate), blocked MT M-tiles per block so activations are read once
// or twice (not Cout/16 times). Channel stride is compile-time (SSP) so the
// per-lane B loads use one base address + immediate offsets. InstanceNorm is
// folded into the consuming GEMMs as a per-(b,ci) column scale + per-(b,co)
// bias correction. Depthwise 3x3x3, window attention, SE and 48-pt DFT line
// passes are dedicated kernels.
//
// Input flattening assumption: recursive insertion order of setup_inputs():
//   d_in[0]=x, then params: cpe1_dw,cpe1_pw,cpe2_dw,cpe2_pw,q_w,q_b,k_w,k_b,
//   v_w,v_b,proj_w,proj_b,fc1_w,fc1_b,mlp_dw,mlp_pw,fc2_w,fc2_b, then per bd
//   layer: real{dw1,pw1,se_w1,se_w2,dw2,pw2}, imag{...}, fus_dw, fus_pw.
// ---------------------------------------------------------------------------

typedef __attribute__((ext_vector_type(16))) _Float16 v16h;
typedef __attribute__((ext_vector_type(8)))  float    v8f;

constexpr int       BB   = 2;
constexpr int       CH0  = 64;
constexpr int       DD   = 48;
constexpr int       SSP  = DD * DD * DD;                 // 110592
constexpr long long VV   = (long long)BB * CH0 * SSP;    // 14155776
constexpr int       NWIN = DD / 4;                       // 12

// ---------------------------------------------------------------------------
// Pointwise conv as GEMM:  Y[b,co,s] = act( sum_ci A'[co,ci] * X[b,ci,s] )
// A'[co,ci] = W[co,ci] * scale_c[ci] * scale_bc[b,ci]   (scales optional)
// 128 threads = 4 waves; block computes (16*MT) Cout x 64 spatial; K step 32.
// B operand: one base pointer per K slab, 16 loads at immediate offsets,
// reused across the MT M-tiles (MT wmma per slab).
// ---------------------------------------------------------------------------
template<int CIN, int MT>
__global__ __launch_bounds__(128)
void pw_gemm_k(const float* __restrict__ X, const float* __restrict__ W,
               const float* __restrict__ bias, const float* __restrict__ scale_c,
               const float* __restrict__ scale_bc, const float* __restrict__ bias_bc,
               const float* __restrict__ resid,
               float* __restrict__ Y, int relu)
{
    __shared__ _Float16 Al[16 * MT * CIN];
    const int b    = blockIdx.z;
    const int cout = gridDim.y * 16 * MT;
    const int coT  = blockIdx.y * 16 * MT;
    const int tid  = threadIdx.x;

    for (int i = tid; i < 16 * MT * CIN; i += 128) {
        int m = i / CIN, kk = i - m * CIN;
        float w = W[(coT + m) * CIN + kk];
        if (scale_c)  w *= scale_c[kk];
        if (scale_bc) w *= scale_bc[b * CIN + kk];
        Al[i] = (_Float16)w;
    }
    __syncthreads();

    const int lane = tid & 31;
    const int wave = tid >> 5;
    const int n    = lane & 15;
    const int g    = lane >> 4;
    const int m    = lane & 15;
    const int s    = blockIdx.x * 64 + wave * 16 + n;
    const float* Xs = X + (long long)b * CIN * SSP + s;

    v8f acc[MT];
#pragma unroll
    for (int t = 0; t < MT; ++t) acc[t] = v8f{};

    for (int k0 = 0; k0 < CIN; k0 += 32) {
        // one base per 16-row slab; row offsets are compile-time immediates
        const float* bp = Xs + (long long)(k0 + (g ? 16 : 0)) * SSP;
        if (k0 + 32 < CIN)   // gfx1250 global_prefetch of next K slab
            __builtin_prefetch((const void*)(bp + 32LL * SSP), 0, 1);
        v16h bv;
#pragma unroll
        for (int v = 0; v < 8; ++v) {   // B 32x16: lanes0-15 K 0-15, lanes16-31 K 16-31
            bv[2 * v]     = (_Float16)bp[(long long)(2 * v) * SSP];
            bv[2 * v + 1] = (_Float16)bp[(long long)(2 * v + 1) * SSP];
        }
#pragma unroll
        for (int t = 0; t < MT; ++t) {
            v16h av;
#pragma unroll
            for (int v = 0; v < 8; ++v) {   // A 16x32: lanes0-15 K 0-15, lanes16-31 K 16-31
                int kk = (g ? 16 : 0) + 2 * v;
                av[2 * v]     = Al[(t * 16 + m) * CIN + k0 + kk];
                av[2 * v + 1] = Al[(t * 16 + m) * CIN + k0 + kk + 1];
            }
            acc[t] = __builtin_amdgcn_wmma_f32_16x16x32_f16(false, av, false, bv,
                                                            (short)0, acc[t], false, false);
        }
    }

    const long long obase = ((long long)b * cout + coT + 8 * g) * SSP + s;
    float*       yp = Y + obase;
    const float* rp = resid ? resid + obase : nullptr;
#pragma unroll
    for (int t = 0; t < MT; ++t) {
#pragma unroll
        for (int r = 0; r < 8; ++r) {       // C/D: M = 8*g + r, N = lane&15
            int   co  = coT + t * 16 + 8 * g + r;
            float val = acc[t][r];
            if (bias)    val += bias[co];
            if (bias_bc) val += bias_bc[b * cout + co];
            if (relu)    val = fmaxf(val, 0.f);
            long long off = (long long)(t * 16 + r) * SSP;
            if (resid)   val += rp[off];
            yp[off] = val;
        }
    }
}

// Fusion GEMM: Cin=192 concatenated from three 64-ch tensors (xb, Re(xs), Im(xs)).
__global__ __launch_bounds__(128)
void pw_gemm_cat3_k(const float* __restrict__ X0, const float* __restrict__ X1,
                    const float* __restrict__ X2, const float* __restrict__ W,
                    const float* __restrict__ scale_c, const float* __restrict__ resid,
                    float* __restrict__ Y)
{
    constexpr int CIN = 192;
    constexpr int MT  = 4;          // all 64 output channels in one block
    __shared__ _Float16 Al[16 * MT * CIN];
    const int b   = blockIdx.z;
    const int tid = threadIdx.x;

    for (int i = tid; i < 16 * MT * CIN; i += 128) {
        int m = i / CIN, kk = i - m * CIN;
        Al[i] = (_Float16)(W[m * CIN + kk] * scale_c[kk]);
    }
    __syncthreads();

    const int lane = tid & 31;
    const int wave = tid >> 5;
    const int n    = lane & 15;
    const int g    = lane >> 4;
    const int m    = lane & 15;
    const int s    = blockIdx.x * 64 + wave * 16 + n;
    const long long bs = (long long)b * 64 * SSP + s;

    v8f acc[MT];
#pragma unroll
    for (int t = 0; t < MT; ++t) acc[t] = v8f{};

#pragma unroll
    for (int k0 = 0; k0 < CIN; k0 += 32) {
        // 16-row slab lies entirely inside one 64-ch concat block
        int cb = k0 + (g ? 16 : 0);
        const float* src = (cb < 64) ? X0 : ((cb < 128) ? X1 : X2);
        const float* bp  = src + bs + (long long)(cb & 63) * SSP;
        v16h bv;
#pragma unroll
        for (int v = 0; v < 8; ++v) {
            bv[2 * v]     = (_Float16)bp[(long long)(2 * v) * SSP];
            bv[2 * v + 1] = (_Float16)bp[(long long)(2 * v + 1) * SSP];
        }
#pragma unroll
        for (int t = 0; t < MT; ++t) {
            v16h av;
#pragma unroll
            for (int v = 0; v < 8; ++v) {
                int kk = (g ? 16 : 0) + 2 * v;
                av[2 * v]     = Al[(t * 16 + m) * CIN + k0 + kk];
                av[2 * v + 1] = Al[(t * 16 + m) * CIN + k0 + kk + 1];
            }
            acc[t] = __builtin_amdgcn_wmma_f32_16x16x32_f16(false, av, false, bv,
                                                            (short)0, acc[t], false, false);
        }
    }

    const long long obase = ((long long)b * 64 + 8 * g) * SSP + s;
    float*       yp = Y + obase;
    const float* rp = resid + obase;
#pragma unroll
    for (int t = 0; t < MT; ++t) {
#pragma unroll
        for (int r = 0; r < 8; ++r) {
            long long off = (long long)(t * 16 + r) * SSP;
            yp[off] = acc[t][r] + rp[off];
        }
    }
}

// ---------------------------------------------------------------------------
// Depthwise 3x3x3 conv, pad=1, bias-free.
// ---------------------------------------------------------------------------
__global__ __launch_bounds__(256)
void dwconv3_k(const float* __restrict__ X, const float* __restrict__ W,
               float* __restrict__ Y, int CH)
{
    long long idx = (long long)blockIdx.x * blockDim.x + threadIdx.x;
    long long total = (long long)BB * CH * SSP;
    if (idx >= total) return;
    int sp = (int)(idx % SSP);
    long long bc = idx / SSP;
    int c  = (int)(bc % CH);
    int z  = sp / (DD * DD);
    int r  = sp - z * DD * DD;
    int y  = r / DD;
    int xw = r - y * DD;
    const float* base = X + bc * SSP;
    const float* wk   = W + c * 27;
    float a = 0.f;
#pragma unroll
    for (int kz = -1; kz <= 1; ++kz) {
        int zz = z + kz; if ((unsigned)zz >= (unsigned)DD) continue;
#pragma unroll
        for (int ky = -1; ky <= 1; ++ky) {
            int yy = y + ky; if ((unsigned)yy >= (unsigned)DD) continue;
#pragma unroll
            for (int kx = -1; kx <= 1; ++kx) {
                int xx = xw + kx; if ((unsigned)xx >= (unsigned)DD) continue;
                a += base[(zz * DD + yy) * DD + xx] * wk[(kz + 1) * 9 + (ky + 1) * 3 + (kx + 1)];
            }
        }
    }
    Y[idx] = a;
}

// ---------------------------------------------------------------------------
// InstanceNorm statistics only: mean and rsqrt(var+eps) per (b,c).
// Normalization itself is folded into the consuming GEMMs.
// ---------------------------------------------------------------------------
__global__ __launch_bounds__(256)
void stats_k(const float* __restrict__ X, float* __restrict__ meanO,
             float* __restrict__ invO)
{
    __shared__ float r1[256], r2[256];
    const long long off = (long long)blockIdx.x * SSP;
    float s = 0.f, s2 = 0.f;
    for (int i = threadIdx.x; i < SSP; i += 256) {
        float v = X[off + i]; s += v; s2 += v * v;
    }
    r1[threadIdx.x] = s; r2[threadIdx.x] = s2;
    __syncthreads();
    for (int st = 128; st > 0; st >>= 1) {
        if (threadIdx.x < st) { r1[threadIdx.x] += r1[threadIdx.x + st];
                                r2[threadIdx.x] += r2[threadIdx.x + st]; }
        __syncthreads();
    }
    if (threadIdx.x == 0) {
        float mean = r1[0] / (float)SSP;
        float var  = r2[0] / (float)SSP - mean * mean;
        meanO[blockIdx.x] = mean;
        invO[blockIdx.x]  = rsqrtf(var + 1e-5f);
    }
}

// bias_bc[b,co] = bias[co] - sum_ci W[co,ci]*inv[b,ci]*mean[b,ci]
__global__ __launch_bounds__(64)
void nbias_k(const float* __restrict__ W, const float* __restrict__ bias,
             const float* __restrict__ mean, const float* __restrict__ inv,
             float* __restrict__ out, int CIN)
{
    __shared__ float r[64];
    int co = blockIdx.x, b = blockIdx.y, tid = threadIdx.x;
    float a = 0.f;
    for (int ci = tid; ci < CIN; ci += 64)
        a += W[co * CIN + ci] * inv[b * CIN + ci] * mean[b * CIN + ci];
    r[tid] = a;
    __syncthreads();
    for (int st = 32; st > 0; st >>= 1) {
        if (tid < st) r[tid] += r[tid + st];
        __syncthreads();
    }
    if (tid == 0) out[b * gridDim.x + co] = (bias ? bias[co] : 0.f) - r[0];
}

// Global average pool per (b,c).
__global__ __launch_bounds__(256)
void mean_k(const float* __restrict__ X, float* __restrict__ out)
{
    __shared__ float r1[256];
    const long long off = (long long)blockIdx.x * SSP;
    float s = 0.f;
    for (int i = threadIdx.x; i < SSP; i += 256) s += X[off + i];
    r1[threadIdx.x] = s;
    __syncthreads();
    for (int st = 128; st > 0; st >>= 1) {
        if (threadIdx.x < st) r1[threadIdx.x] += r1[threadIdx.x + st];
        __syncthreads();
    }
    if (threadIdx.x == 0) out[blockIdx.x] = r1[0] / (float)SSP;
}

// SE gate: ys[b,c] = sigmoid( relu(mean @ w1^T) @ w2^T ). One block per b, 64 thr.
__global__ __launch_bounds__(64)
void se_mlp_k(const float* __restrict__ mean, const float* __restrict__ w1,
              const float* __restrict__ w2, float* __restrict__ ys)
{
    __shared__ float m[64];
    __shared__ float t[16];
    int b = blockIdx.x, tid = threadIdx.x;
    m[tid] = mean[b * 64 + tid];
    __syncthreads();
    if (tid < 16) {
        float a = 0.f;
        for (int j = 0; j < 64; ++j) a += w1[tid * 64 + j] * m[j];
        t[tid] = fmaxf(a, 0.f);
    }
    __syncthreads();
    float a = 0.f;
    for (int j = 0; j < 16; ++j) a += w2[tid * 16 + j] * t[j];
    ys[b * 64 + tid] = 1.f / (1.f + expf(-a));
}

// ---------------------------------------------------------------------------
// 4x4x4 window attention: one block per window, all 4 heads.
// q/k/v staged as f16 in LDS (40.3 KB total incl. f32 prob tile).
// ---------------------------------------------------------------------------
__global__ __launch_bounds__(256)
void attn_k(const float* __restrict__ Q, const float* __restrict__ K,
            const float* __restrict__ V, float* __restrict__ O)
{
    extern __shared__ char smc[];
    _Float16* qs = (_Float16*)smc;            // 64ch x 64pos f16
    _Float16* ks = qs + 4096;
    _Float16* vs = ks + 4096;
    float*    at = (float*)(smc + 24576);     // 64x64 logits/probs f32
    int*    sIdx = (int*)(smc + 24576 + 16384);

    int win = blockIdx.x;
    int b   = win / (NWIN * NWIN * NWIN);
    int r   = win - b * (NWIN * NWIN * NWIN);
    int wd  = r / (NWIN * NWIN); r -= wd * NWIN * NWIN;
    int wh  = r / NWIN;
    int ww  = r - wh * NWIN;
    int tid = threadIdx.x;

    if (tid < 64) {
        int dz = tid >> 4, dy = (tid >> 2) & 3, dx = tid & 3;
        sIdx[tid] = ((wd * 4 + dz) * DD + wh * 4 + dy) * DD + ww * 4 + dx;
    }
    __syncthreads();
    for (int i = tid; i < 4096; i += 256) {
        int c = i >> 6, t = i & 63;
        long long gi = ((long long)b * CH0 + c) * SSP + sIdx[t];
        qs[i] = (_Float16)Q[gi]; ks[i] = (_Float16)K[gi]; vs[i] = (_Float16)V[gi];
    }
    __syncthreads();
    for (int h = 0; h < 4; ++h) {
        for (int i = tid; i < 4096; i += 256) {     // logits
            int t = i >> 6, sj = i & 63;
            float a = 0.f;
#pragma unroll
            for (int c = 0; c < 16; ++c)
                a += (float)qs[(h * 16 + c) * 64 + t] * (float)ks[(h * 16 + c) * 64 + sj];
            at[t * 64 + sj] = a * 0.25f;            // hd^-0.5, hd=16
        }
        __syncthreads();
        if (tid < 64) {                             // softmax over sj
            float mx = -3.4e38f;
            for (int j = 0; j < 64; ++j) mx = fmaxf(mx, at[tid * 64 + j]);
            float sum = 0.f;
            for (int j = 0; j < 64; ++j) { float e = expf(at[tid * 64 + j] - mx);
                                           at[tid * 64 + j] = e; sum += e; }
            float inv = 1.f / sum;
            for (int j = 0; j < 64; ++j) at[tid * 64 + j] *= inv;
        }
        __syncthreads();
        for (int i = tid; i < 1024; i += 256) {     // out[c,t] = sum_s v[c,s]*p[t,s]
            int c = i >> 6, t = i & 63;
            float a = 0.f;
            for (int sj = 0; sj < 64; ++sj)
                a += (float)vs[(h * 16 + c) * 64 + sj] * at[t * 64 + sj];
            O[((long long)b * CH0 + h * 16 + c) * SSP + sIdx[t]] = a;
        }
        __syncthreads();
    }
}

// ---------------------------------------------------------------------------
// 48-pt DFT along one axis (stride-parameterized). Thread owns one line,
// staged in LDS rows padded to 49 floats (bank-conflict free). 128 thr/block
// keeps LDS at 50.5 KB. In-place safe (thread owns its whole line).
// ---------------------------------------------------------------------------
__global__ __launch_bounds__(128)
void fft48_k(const float* inRe, const float* inIm,
             float* outRe, float* outIm,
             int stride, int nlines, int inverse, float scale)
{
    extern __shared__ float sm[];
    float* twr = sm;                 // 48
    float* twi = sm + 48;            // 48
    float* lre = sm + 96;            // 128*49
    float* lim = sm + 96 + 128 * 49;

    int tid = threadIdx.x;
    if (tid < 48) {
        float ang = 6.283185307179586f * (float)tid / 48.f;
        twr[tid] = cosf(ang);
        float sn = sinf(ang);
        twi[tid] = inverse ? sn : -sn;
    }
    __syncthreads();

    int l = blockIdx.x * 128 + tid;
    if (l >= nlines) return;
    int outer = l / stride, inner = l - outer * stride;
    long long base = (long long)outer * 48 * stride + inner;
    int row = tid * 49;
    for (int j = 0; j < 48; ++j) {
        long long gi = base + (long long)j * stride;
        lre[row + j] = inRe[gi];
        lim[row + j] = inIm ? inIm[gi] : 0.f;
    }
    for (int k = 0; k < 48; ++k) {
        float ar = 0.f, ai = 0.f;
        int p = 0;
        for (int j = 0; j < 48; ++j) {
            float xr = lre[row + j], xi = lim[row + j];
            float tr = twr[p], ti = twi[p];
            ar += xr * tr - xi * ti;
            ai += xr * ti + xi * tr;
            p += k; if (p >= 48) p -= 48;
        }
        long long gi = base + (long long)k * stride;
        outRe[gi] = ar * scale;
        outIm[gi] = ai * scale;
    }
}

__global__ __launch_bounds__(256)
void add_k(const float* __restrict__ a, const float* __restrict__ b,
           float* __restrict__ o, long long n)
{
    long long i = (long long)blockIdx.x * blockDim.x + threadIdx.x;
    if (i < n) o[i] = a[i] + b[i];
}

// ---------------------------------------------------------------------------
extern "C" void kernel_launch(void* const* d_in, const int* in_sizes, int n_in,
                              void* d_out, int out_size, void* d_ws, size_t ws_size,
                              hipStream_t stream)
{
    (void)in_sizes; (void)n_in; (void)out_size; (void)ws_size;
    auto P = [&](int i) { return (const float*)d_in[i]; };

    const float* x       = P(0);
    const float* cpe1_dw = P(1);  const float* cpe1_pw = P(2);
    const float* cpe2_dw = P(3);  const float* cpe2_pw = P(4);
    const float* q_w = P(5);  const float* q_b = P(6);
    const float* k_w = P(7);  const float* k_b = P(8);
    const float* v_w = P(9);  const float* v_b = P(10);
    const float* proj_w = P(11); const float* proj_b = P(12);
    const float* fc1_w = P(13); const float* fc1_b = P(14);
    const float* mlp_dw = P(15); const float* mlp_pw = P(16);
    const float* fc2_w = P(17); const float* fc2_b = P(18);

    float* WSf = (float*)d_ws;
    auto wsb = [&](int i) { return WSf + (long long)i * VV; };
    float* t1   = wsb(0);
    float* x1   = wsb(1);
    float* x2   = wsb(2);
    float* qb   = wsb(3);            // later x3
    float* kb   = wsb(4);
    float* vb   = wsb(5);            // later x4
    float* xbA  = wsb(6);            // bidomain xb ping
    float* xbB  = wsb(7);            // bidomain xb pong
    float* mlpA = wsb(8);            // 4V  (8..11)
    float* mlpB = wsb(12);           // 4V  (12..15)
    // after MLP, regions 8..15 are recycled:
    float* xfr  = wsb(8);  float* xfi  = wsb(9);
    float* xsr  = wsb(10); float* xsi  = wsb(11);
    float* bufA = wsb(12); float* bufR = wsb(13); float* bufI = wsb(14);
    float* nMean   = wsb(15);
    float* nInv    = wsb(15) + 128;
    float* seMean  = wsb(15) + 256;
    float* seScale = wsb(15) + 384;
    float* nBias   = wsb(15) + 512;  // up to B*256

    dim3 blk(128);
    const int GX = SSP / 64;         // 1728 spatial tiles
    auto pw64 = [&](const float* X, const float* W, const float* bias,
                    const float* sc, const float* sbc, const float* bbc,
                    const float* resid, float* Y, int relu) {
        dim3 g(GX, 1, BB);           // COUT=64, MT=4
        pw_gemm_k<64, 4><<<g, blk, 0, stream>>>(X, W, bias, sc, sbc, bbc, resid, Y, relu);
    };

    // ---- CPE1: x1 = x + PW(DW3(x)) ----
    dwconv3_k<<<(unsigned)(VV / 256), 256, 0, stream>>>(x, cpe1_dw, t1, CH0);
    pw64(t1, cpe1_pw, nullptr, nullptr, nullptr, nullptr, x, x1, 0);

    // ---- Window attention (inorm folded into q/k/v GEMMs) ----
    stats_k<<<BB * CH0, 256, 0, stream>>>(x1, nMean, nInv);
    nbias_k<<<dim3(64, BB), 64, 0, stream>>>(q_w, q_b, nMean, nInv, nBias, 64);
    pw64(x1, q_w, nullptr, nullptr, nInv, nBias, nullptr, qb, 0);
    nbias_k<<<dim3(64, BB), 64, 0, stream>>>(k_w, k_b, nMean, nInv, nBias, 64);
    pw64(x1, k_w, nullptr, nullptr, nInv, nBias, nullptr, kb, 0);
    nbias_k<<<dim3(64, BB), 64, 0, stream>>>(v_w, v_b, nMean, nInv, nBias, 64);
    pw64(x1, v_w, nullptr, nullptr, nInv, nBias, nullptr, vb, 0);
    float* att = t1;  // reuse
    {
        size_t shm = 24576 + 16384 + 256;   // f16 q/k/v + f32 probs + sIdx
        attn_k<<<BB * NWIN * NWIN * NWIN, 256, shm, stream>>>(qb, kb, vb, att);
    }
    pw64(att, proj_w, proj_b, nullptr, nullptr, nullptr, x1, x2, 0);

    // ---- CPE2: x3 = x2 + PW(DW3(x2)) ----
    float* x3 = qb;
    dwconv3_k<<<(unsigned)(VV / 256), 256, 0, stream>>>(x2, cpe2_dw, t1, CH0);
    pw64(t1, cpe2_pw, nullptr, nullptr, nullptr, nullptr, x2, x3, 0);

    // ---- MLP: x4 = x3 + FC2(relu(PW(DW3(FC1(inorm(x3)))))) ----
    stats_k<<<BB * CH0, 256, 0, stream>>>(x3, nMean, nInv);
    nbias_k<<<dim3(256, BB), 64, 0, stream>>>(fc1_w, fc1_b, nMean, nInv, nBias, 64);
    {   // fc1: 64 -> 256, MT=8 (128 co/block)
        dim3 g(GX, 2, BB);
        pw_gemm_k<64, 8><<<g, blk, 0, stream>>>(x3, fc1_w, nullptr, nullptr, nInv, nBias,
                                                nullptr, mlpA, 0);
    }
    dwconv3_k<<<(unsigned)((VV * 4) / 256), 256, 0, stream>>>(mlpA, mlp_dw, mlpB, 256);
    {   // mlp_pw: 256 -> 256, relu, MT=8
        dim3 g(GX, 2, BB);
        pw_gemm_k<256, 8><<<g, blk, 0, stream>>>(mlpB, mlp_pw, nullptr, nullptr, nullptr,
                                                 nullptr, nullptr, mlpA, 1);
    }
    float* x4 = vb;
    {   // fc2: 256 -> 64, MT=4, residual x3
        dim3 g(GX, 1, BB);
        pw_gemm_k<256, 4><<<g, blk, 0, stream>>>(mlpA, fc2_w, fc2_b, nullptr, nullptr,
                                                 nullptr, x3, x4, 0);
    }

    // ---- Bidomain: xb = x4; xf = fftn(x4) ----
    hipMemcpyAsync(xbA, x4, (size_t)VV * sizeof(float), hipMemcpyDeviceToDevice, stream);
    const int nl = (int)(VV / 48);
    dim3 fg(nl / 128);
    size_t fsh = (size_t)(96 + 2 * 128 * 49) * sizeof(float);
    fft48_k<<<fg, 128, fsh, stream>>>(x4, nullptr, xfr, xfi, 1,    nl, 0, 1.f);
    fft48_k<<<fg, 128, fsh, stream>>>(xfr, xfi,    xfr, xfi, 48,   nl, 0, 1.f);
    fft48_k<<<fg, 128, fsh, stream>>>(xfr, xfi,    xfr, xfi, 2304, nl, 0, 1.f);

    float* curR = xfr; float* curI = xfi;
    float* altR = bufR; float* altI = bufI;
    const float is = 1.f / 48.f;
    for (int L = 0; L < 2; ++L) {
        int pb = 19 + L * 14;
        // real branch: relu(PW1(dw1*Re)) -> SE -> PW2(dw2*se*h)
        pw64(curR, P(pb + 1), nullptr, P(pb + 0), nullptr, nullptr, nullptr, bufA, 1);
        mean_k<<<BB * CH0, 256, 0, stream>>>(bufA, seMean);
        se_mlp_k<<<BB, 64, 0, stream>>>(seMean, P(pb + 2), P(pb + 3), seScale);
        pw64(bufA, P(pb + 5), nullptr, P(pb + 4), seScale, nullptr, nullptr, altR, 0);
        // imag branch
        pw64(curI, P(pb + 7), nullptr, P(pb + 6), nullptr, nullptr, nullptr, bufA, 1);
        mean_k<<<BB * CH0, 256, 0, stream>>>(bufA, seMean);
        se_mlp_k<<<BB, 64, 0, stream>>>(seMean, P(pb + 8), P(pb + 9), seScale);
        pw64(bufA, P(pb + 11), nullptr, P(pb + 10), seScale, nullptr, nullptr, altI, 0);
        // xs = ifftn(xf)   (1/48 per axis pass)
        fft48_k<<<fg, 128, fsh, stream>>>(altR, altI, xsr, xsi, 1,    nl, 1, is);
        fft48_k<<<fg, 128, fsh, stream>>>(xsr, xsi,  xsr, xsi, 48,   nl, 1, is);
        fft48_k<<<fg, 128, fsh, stream>>>(xsr, xsi,  xsr, xsi, 2304, nl, 1, is);
        // fusion: xb' = xb + PW(fus_dw * cat[xb, Re(xs), Im(xs)])
        {
            dim3 g(GX, 1, BB);
            pw_gemm_cat3_k<<<g, blk, 0, stream>>>(xbA, xsr, xsi, P(pb + 13), P(pb + 12),
                                                  xbA, xbB);
        }
        { float* t = xbA; xbA = xbB; xbB = t; }
        { float* t = curR; curR = altR; altR = t; }
        { float* t = curI; curI = altI; altI = t; }
    }

    // ---- out = x4 + xb ----
    add_k<<<(unsigned)(VV / 256), 256, 0, stream>>>(x4, xbA, (float*)d_out, VV);
}
